// QuantizedLinear_57002805952633
// MI455X (gfx1250) — matrix operand order repaired and checked by reading, not compiled, not run on hardware
//
#include <hip/hip_runtime.h>

typedef __attribute__((ext_vector_type(16))) _Float16 v16h;
typedef __attribute__((ext_vector_type(8)))  float    v8f;
typedef __attribute__((ext_vector_type(8)))  _Float16 h8;
typedef __attribute__((ext_vector_type(4)))  _Float16 h4;
typedef __attribute__((ext_vector_type(4)))  float    f4;
typedef __attribute__((ext_vector_type(4)))  int      i4;

#define BM 128
#define BN 128
#define BK 64
#define LDSS (BK + 8)   // padded LDS row stride in halves (72 -> conflict-free b128 reads)

__global__ __launch_bounds__(256) void qlinear_wmma(
    const float* __restrict__ x,      // [M, K] f32
    const int*   __restrict__ wq,     // [N, K] int32 in [0,256)
    const float* __restrict__ scale,  // [N*K/128]
    const float* __restrict__ zp,     // [N*K/128]
    const float* __restrict__ bias,   // [N]
    float* __restrict__ out,          // [M, N]
    int M, int N, int K)
{
    // double-buffered tiles: 2 x (128*72*2B) x 2 = 72 KB of the 320 KB WGP LDS
    __shared__ _Float16 xs[2][BM * LDSS];   // x tile, f16
    __shared__ _Float16 ws[2][BN * LDSS];   // dequantized w tile, f16

    const int tid = threadIdx.x;
    // m-tile on the FAST grid dim: x stays L2-resident across the whole dispatch
    const int m0 = blockIdx.x * BM;
    const int n0 = blockIdx.y * BN;

    // ---- tile loader mapping: thread t -> row t/2, K-half (t&1)*32, 8x 16B loads
    const int lr = tid >> 1;
    const int lk = (tid & 1) << 5;                   // 0 or 32

    const float* xg = x  + (size_t)(m0 + lr) * K + lk;
    const int*   wg = wq + (size_t)(n0 + lr) * K + lk;
    const int    wgrow = (n0 + lr) * (K >> 7);       // first group index of this w row

    // ---- wave / fragment mapping (wave32)
    const int wave = tid >> 5;                       // 0..7
    const int lane = tid & 31;
    const int hi16 = lane >> 4;                      // 0 or 1 (half-wave)
    const int l15  = lane & 15;
    const int wm   = (wave >> 2) << 6;               // 0 or 64
    const int wn   = (wave & 3) << 5;                // 0,32,64,96

    v8f acc[4][2] = {};

    // register prefetch of chunk 0
    f4 xr[8]; i4 wr[8];
#pragma unroll
    for (int j = 0; j < 8; ++j) {
        xr[j] = *(const f4*)(xg + 4 * j);
        wr[j] = *(const i4*)(wg + 4 * j);
    }

    int p = 0;
    for (int kc = 0; kc < K; kc += BK) {
        // group-constant dequant params for this 64-wide K chunk (G=128 >= BK)
        const int   g  = wgrow + (kc >> 7);
        const float sc = scale[g];
        const float z  = zp[g];

        _Float16* xdst = &xs[p][lr * LDSS + lk];
        _Float16* wdst = &ws[p][lr * LDSS + lk];
#pragma unroll
        for (int j = 0; j < 8; ++j) {
            h4 xh = __builtin_convertvector(xr[j], h4);
            f4 wf = (__builtin_convertvector(wr[j], f4) - z) * sc;  // dequantize
            h4 wh = __builtin_convertvector(wf, h4);
            *(h4*)(xdst + 4 * j) = xh;
            *(h4*)(wdst + 4 * j) = wh;
        }

        // issue next chunk's global loads before the barrier so they overlap WMMA
        if (kc + BK < K) {
            xg += BK; wg += BK;
#pragma unroll
            for (int j = 0; j < 8; ++j) {
                xr[j] = *(const f4*)(xg + 4 * j);
                wr[j] = *(const i4*)(wg + 4 * j);
            }
        }

        // single barrier per chunk: separates this chunk's LDS stores from its reads,
        // and (via the alternating buffer) last chunk's reads from next chunk's stores
        __syncthreads();

#pragma unroll
        for (int ks = 0; ks < BK; ks += 32) {
            union { v16h v; h8 h[2]; } a[4], b[2];
            // A 16x32 f16 layout: lanes 0-15 row=l15 K{0..7,16..23}; lanes 16-31 K{8..15,24..31}
#pragma unroll
            for (int mi = 0; mi < 4; ++mi) {
                const _Float16* q = &xs[p][(wm + mi * 16 + l15) * LDSS + ks + hi16 * 8];
                a[mi].h[0] = *(const h8*)q;
                a[mi].h[1] = *(const h8*)(q + 16);
            }
            // B 32x16 f16 layout: lane holds column l15, K{0..7,16..23} (lanes 0-15); K{8..15,24..31} (lanes 16-31)
#pragma unroll
            for (int ni = 0; ni < 2; ++ni) {
                const _Float16* q = &ws[p][(wn + ni * 16 + l15) * LDSS + ks + hi16 * 8];
                b[ni].h[0] = *(const h8*)q;
                b[ni].h[1] = *(const h8*)(q + 16);
            }
#pragma unroll
            for (int mi = 0; mi < 4; ++mi)
#pragma unroll
                for (int ni = 0; ni < 2; ++ni)
                    acc[mi][ni] = __builtin_amdgcn_wmma_f32_16x16x32_f16(
                        false, a[mi].v, false, b[ni].v,
                        (short)0, acc[mi][ni], false, false);
        }
        p ^= 1;
    }

    // epilogue: D layout -> VGPR r: M = r + 8*hi16, N = l15; add bias, coalesced b32 stores
#pragma unroll
    for (int ni = 0; ni < 2; ++ni) {
        const int   n  = n0 + wn + ni * 16 + l15;
        const float bv = bias[n];
#pragma unroll
        for (int mi = 0; mi < 4; ++mi) {
            const int mb = m0 + wm + mi * 16 + hi16 * 8;
#pragma unroll
            for (int r = 0; r < 8; ++r)
                out[(size_t)(mb + r) * N + n] = acc[mi][ni][r] + bv;
        }
    }
}

extern "C" void kernel_launch(void* const* d_in, const int* in_sizes, int n_in,
                              void* d_out, int out_size, void* d_ws, size_t ws_size,
                              hipStream_t stream) {
    const float* x     = (const float*)d_in[0];
    const int*   wq    = (const int*)d_in[1];
    const float* scale = (const float*)d_in[2];
    const float* zp    = (const float*)d_in[3];
    const float* bias  = (const float*)d_in[4];
    float* out = (float*)d_out;

    const int N = in_sizes[4];            // 16384
    const int K = in_sizes[1] / N;        // 4096
    const int M = in_sizes[0] / K;        // 8192

    dim3 grid(M / BM, N / BN);            // m fast -> x L2-resident, w streamed once
    qlinear_wmma<<<grid, 256, 0, stream>>>(x, wq, scale, zp, bias, out, M, N, K);
}
